// NGU_72138270704384
// MI455X (gfx1250) — matrix operand order repaired and checked by reading, not compiled, not run on hardware
//
#include <hip/hip_runtime.h>
#include <math.h>

#define N_PTS   8192
#define SDIM    512
#define HID     128
#define DEMB    128
#define KSEL    10
#define TOPN    (KSEL + 1)       // 11: k nearest + self
#define RB      64               // rows per block (knn kernel)
#define CB      64               // column tile
#define LSTR    136              // LDS row stride in halves (pad vs bank conflicts)
#define GSTR    68               // LDS row stride of Gram tile in floats
#define WSTR    40               // LDS stride of W1 K-slice (halves)
#define HSTR    136              // LDS stride of hidden activations (halves)
#define XSTR    132              // LDS stride of x staging (floats)

typedef _Float16 v8h  __attribute__((ext_vector_type(8)));
typedef _Float16 v16h __attribute__((ext_vector_type(16)));
typedef float    v8f  __attribute__((ext_vector_type(8)));
typedef int      v4i  __attribute__((vector_size(16)));

// pointer-to-v4i in explicit address spaces (for the async-LDS builtin)
typedef __attribute__((address_space(1))) v4i* gas_v4i_p;   // global
typedef __attribute__((address_space(3))) v4i* las_v4i_p;   // LDS

#if __has_builtin(__builtin_amdgcn_global_load_async_to_lds_b128)
#define HAVE_ASYNC 1
#else
#define HAVE_ASYNC 0
#endif

// 16B global -> LDS copy: async (ASYNCcnt) when available, sync fallback.
__device__ __forceinline__ void cp16_g2l(const _Float16* g, _Float16* l) {
#if HAVE_ASYNC
  __builtin_amdgcn_global_load_async_to_lds_b128(
      (gas_v4i_p)(const void*)g, (las_v4i_p)(void*)l, 0, 0);
#else
  *(uint4*)l = *(const uint4*)g;
#endif
}

__device__ __forceinline__ void wait_async() {
#if HAVE_ASYNC
#if __has_builtin(__builtin_amdgcn_s_wait_asynccnt)
  __builtin_amdgcn_s_wait_asynccnt(0);
#else
  asm volatile("s_wait_asynccnt 0x0" ::: "memory");
#endif
#endif
}

// Load a 16-element f16 fragment as two aligned 16B reads.
__device__ __forceinline__ v16h frag16(const _Float16* p, int step) {
  v8h a = *(const v8h*)p;
  v8h b = *(const v8h*)(p + step);
  return __builtin_shufflevector(a, b, 0,1,2,3,4,5,6,7,8,9,10,11,12,13,14,15);
}

#define WMMA_F16(A, B, C) \
  __builtin_amdgcn_wmma_f32_16x16x32_f16(false, (A), false, (B), (short)0, (C), false, false)

// -------------------- kernel 0: fp32 -> f16 hi/lo split --------------------
__global__ __launch_bounds__(256) void split_kernel(
    const float* __restrict__ src, _Float16* __restrict__ h,
    _Float16* __restrict__ l, int n)
{
  const int i = blockIdx.x * 256 + threadIdx.x;
  if (i < n) {
    float v = src[i];
    _Float16 hi = (_Float16)v;                 // RTNE
    h[i] = hi;
    l[i] = (_Float16)(v - (float)hi);          // ~11 extra mantissa bits
  }
}

// -------------------- kernel 1: WMMA MLP embedding -------------------------
// x = relu(s @ W1^T + b1) @ W2^T + b2, all GEMMs via f16 hi/lo split WMMA.
// 64 rows/block, 8 waves: wave = (rt 0..3) x (ng 0..1), 4 col 16-tiles each.
__global__ __launch_bounds__(256) void mlp_kernel(
    const _Float16* __restrict__ sh,  const _Float16* __restrict__ sl,
    const _Float16* __restrict__ w1h, const _Float16* __restrict__ w1l,
    const _Float16* __restrict__ w2h, const _Float16* __restrict__ w2l,
    const float* __restrict__ b1, const float* __restrict__ b2,
    _Float16* __restrict__ xh, _Float16* __restrict__ xl,
    float* __restrict__ sq)
{
  __shared__ __align__(32) unsigned char smem[76800];
  _Float16* Wsh = (_Float16*)smem;                 // [2][128*WSTR] dbl-buffered
  _Float16* Wsl = (_Float16*)(smem + 20480);       // [2][128*WSTR]
  float*    xs  = (float*)smem;                    // alias (dead Ws): [64*XSTR]
  _Float16* Hh  = (_Float16*)(smem + 40960);       // [64*HSTR]
  _Float16* Hl  = (_Float16*)(smem + 58368);       // [64*HSTR]
  float*    prt = (float*)(smem + 75776);          // [64*4] partial row norms

  const int tid  = threadIdx.x;
  const int lane = tid & 31;
  const int wave = tid >> 5;
  const int rt   = wave >> 1;           // row 16-tile (0..3)
  const int ng   = wave & 1;            // col-tile group: tiles ng*4 .. ng*4+3
  const int m    = lane & 15;
  const int hl   = lane >> 4;
  const int rowBase = blockIdx.x * 64;

  // async-stage one 32-wide K-slice of W1 (hi+lo) into LDS buffer `buf`
  auto stage_w = [&](int buf, int kk) {
    const int k0 = kk * 32;
    _Float16* dh = Wsh + buf * (128 * WSTR);
    _Float16* dl = Wsl + buf * (128 * WSTR);
#pragma unroll
    for (int ii = 0; ii < 2; ++ii) {            // 512 16B chunks / 256 threads
      int i = tid + ii * 256;
      int n = i >> 2, c = i & 3;                // 4 chunks of 8 halves per row
      cp16_g2l(w1h + n * SDIM + k0 + c * 8, dh + n * WSTR + c * 8);
      cp16_g2l(w1l + n * SDIM + k0 + c * 8, dl + n * WSTR + c * 8);
    }
  };

  // ---------------- layer 1: h = relu(s @ W1^T + b1) ----------------
  v8f acc[4] = {{}, {}, {}, {}};
  stage_w(0, 0);
  wait_async();
  __syncthreads();
  int cur = 0;
#pragma unroll 1
  for (int kk = 0; kk < SDIM / 32; ++kk) {
    if (kk + 1 < SDIM / 32) stage_w(cur ^ 1, kk + 1);   // prefetch next slice
    const int k0 = kk * 32;
    v16h a_h = frag16(sh + (size_t)(rowBase + rt * 16 + m) * SDIM + k0 + hl * 8, 16);
    v16h a_l = frag16(sl + (size_t)(rowBase + rt * 16 + m) * SDIM + k0 + hl * 8, 16);
    const _Float16* wh = Wsh + cur * (128 * WSTR);
    const _Float16* wl = Wsl + cur * (128 * WSTR);
#pragma unroll
    for (int t = 0; t < 4; ++t) {
      const int n0 = (ng * 4 + t) * 16 + m;
      v16h b_h = frag16(wh + n0 * WSTR + hl * 8, 16);
      v16h b_l = frag16(wl + n0 * WSTR + hl * 8, 16);
      acc[t] = WMMA_F16(a_l, b_h, acc[t]);
      acc[t] = WMMA_F16(a_h, b_l, acc[t]);
      acc[t] = WMMA_F16(a_h, b_h, acc[t]);
    }
    wait_async();
    __syncthreads();
    cur ^= 1;
  }

  // bias + relu, re-split to f16 hi/lo in LDS for layer 2
#pragma unroll
  for (int t = 0; t < 4; ++t) {
    const int col = (ng * 4 + t) * 16 + m;
    const float bv = b1[col];
#pragma unroll
    for (int r = 0; r < 8; ++r) {
      float v = fmaxf(acc[t][r] + bv, 0.0f);
      _Float16 hi = (_Float16)v;
      const int row = rt * 16 + hl * 8 + r;
      Hh[row * HSTR + col] = hi;
      Hl[row * HSTR + col] = (_Float16)(v - (float)hi);
    }
  }
  __syncthreads();

  // ---------------- layer 2: x = h @ W2^T + b2 ----------------
  v8f acc2[4] = {{}, {}, {}, {}};
#pragma unroll
  for (int kk = 0; kk < HID / 32; ++kk) {
    const int k0 = kk * 32;
    v16h a_h = frag16(Hh + (rt * 16 + m) * HSTR + k0 + hl * 8, 16);
    v16h a_l = frag16(Hl + (rt * 16 + m) * HSTR + k0 + hl * 8, 16);
#pragma unroll
    for (int t = 0; t < 4; ++t) {
      const int n0 = (ng * 4 + t) * 16 + m;
      v16h b_h = frag16(w2h + n0 * HID + k0 + hl * 8, 16);   // W2 from L2
      v16h b_l = frag16(w2l + n0 * HID + k0 + hl * 8, 16);
      acc2[t] = WMMA_F16(a_l, b_h, acc2[t]);
      acc2[t] = WMMA_F16(a_h, b_l, acc2[t]);
      acc2[t] = WMMA_F16(a_h, b_h, acc2[t]);
    }
  }

  // stage x tile (fp32) into LDS (aliases dead W-slice buffers)
#pragma unroll
  for (int t = 0; t < 4; ++t) {
    const int col = (ng * 4 + t) * 16 + m;
    const float bv = b2[col];
#pragma unroll
    for (int r = 0; r < 8; ++r)
      xs[(rt * 16 + hl * 8 + r) * XSTR + col] = acc2[t][r] + bv;
  }
  __syncthreads();

  // coalesced xh/xl emission
  for (int i = tid; i < 64 * DEMB; i += 256) {
    int row = i >> 7, col = i & 127;
    float v = xs[row * XSTR + col];
    _Float16 hi = (_Float16)v;
    xh[(size_t)(rowBase + row) * DEMB + col] = hi;
    xl[(size_t)(rowBase + row) * DEMB + col] = (_Float16)(v - (float)hi);
  }
  // fp32 row norms, fixed-order reduction (deterministic)
  {
    const int row = tid >> 2, q = tid & 3;
    float p = 0.0f;
#pragma unroll
    for (int c = 0; c < 32; ++c) {
      float v = xs[row * XSTR + q * 32 + c];
      p = fmaf(v, v, p);
    }
    prt[row * 4 + q] = p;
  }
  __syncthreads();
  if (tid < 64)
    sq[rowBase + tid] = (prt[tid * 4 + 0] + prt[tid * 4 + 1]) +
                        (prt[tid * 4 + 2] + prt[tid * 4 + 3]);
}

// -------------------- fused Gram + kNN top-11 ------------------------------
__device__ __forceinline__ void insert11(float (&b)[TOPN], float d) {
  if (d < b[TOPN - 1]) {
#pragma unroll
    for (int q = 0; q < TOPN; ++q) {      // branchless sorted insert (ascending)
      float mn = fminf(b[q], d);
      float mx = fmaxf(b[q], d);
      b[q] = mn; d = mx;
    }
  }
}

__global__ __launch_bounds__(256) void knn_kernel(
    const _Float16* __restrict__ xh, const _Float16* __restrict__ xl,
    const float* __restrict__ sq, float* __restrict__ sumk)
{
  __shared__ __align__(32) _Float16 Bh[2 * CB * LSTR];   // double-buffered
  __shared__ __align__(32) _Float16 Bl[2 * CB * LSTR];
  __shared__ __align__(16) float    G[RB * GSTR];
  __shared__ float sqjA[2 * CB];

  const int tid  = threadIdx.x;
  const int lane = tid & 31;           // wave32
  const int wave = tid >> 5;           // 8 waves
  const int rt   = wave >> 1;          // row 16-tile   (0..3)
  const int ct0  = (wave & 1) * 2;     // first of two col 16-tiles (0 or 2)
  const int rowBase = blockIdx.x * RB;
  const int m  = lane & 15;
  const int hl = lane >> 4;

  // ---- A fragments (this wave's fixed 16-row stripe) live in registers ----
  v16h a_h[4], a_l[4];
  {
    const _Float16* pa = xh + (size_t)(rowBase + rt * 16 + m) * DEMB + hl * 8;
    const _Float16* pb = xl + (size_t)(rowBase + rt * 16 + m) * DEMB + hl * 8;
#pragma unroll
    for (int kk = 0; kk < 4; ++kk) {
      a_h[kk] = frag16(pa + kk * 32, 16);
      a_l[kk] = frag16(pb + kk * 32, 16);
    }
  }

  const int selRow = tid >> 2;         // 0..63, 4 threads per row
  const int selCol = tid & 3;
  const float sqi = sq[rowBase + selRow];
  float best[TOPN];
#pragma unroll
  for (int q = 0; q < TOPN; ++q) best[q] = 3.0e38f;

  // ---- async staging of a 64-row column tile (hi+lo) into LDS buffer ----
  auto stage_tile = [&](int buf, int jt) {
    _Float16* bh = Bh + buf * (CB * LSTR);
    _Float16* bl = Bl + buf * (CB * LSTR);
    const _Float16* gh = xh + (size_t)jt * DEMB;
    const _Float16* gl = xl + (size_t)jt * DEMB;
#pragma unroll
    for (int ii = 0; ii < (CB * 16) / 256; ++ii) {   // 16B chunks, 4/thread/array
      int i = tid + ii * 256;
      int r = i >> 4, c = i & 15;
      cp16_g2l(gh + i * 8, bh + r * LSTR + c * 8);
      cp16_g2l(gl + i * 8, bl + r * LSTR + c * 8);
    }
    if (tid < CB) sqjA[buf * CB + tid] = sq[jt + tid];
  };

  stage_tile(0, 0);
  wait_async();
  __syncthreads();

  int cur = 0;
  for (int jt = 0; jt < N_PTS; jt += CB) {
    if (jt + CB < N_PTS) stage_tile(cur ^ 1, jt + CB);   // prefetch next tile

    const _Float16* bh = Bh + cur * (CB * LSTR);
    const _Float16* bl = Bl + cur * (CB * LSTR);

    // ---- two 16x16 Gram tiles per wave, f16 hi/lo split (~fp32 accuracy) ----
    v8f acc0 = {}, acc1 = {};
#pragma unroll
    for (int kk = 0; kk < 4; ++kk) {
      const int k0 = kk * 32;
      v16h b_h0 = frag16(bh + ((ct0    ) * 16 + m) * LSTR + k0 + hl * 8, 16);
      v16h b_l0 = frag16(bl + ((ct0    ) * 16 + m) * LSTR + k0 + hl * 8, 16);
      v16h b_h1 = frag16(bh + ((ct0 + 1) * 16 + m) * LSTR + k0 + hl * 8, 16);
      v16h b_l1 = frag16(bl + ((ct0 + 1) * 16 + m) * LSTR + k0 + hl * 8, 16);
      acc0 = WMMA_F16(a_l[kk], b_h0, acc0);
      acc0 = WMMA_F16(a_h[kk], b_l0, acc0);
      acc0 = WMMA_F16(a_h[kk], b_h0, acc0);
      acc1 = WMMA_F16(a_l[kk], b_h1, acc1);
      acc1 = WMMA_F16(a_h[kk], b_l1, acc1);
      acc1 = WMMA_F16(a_h[kk], b_h1, acc1);
    }
    // C/D layout: VGPR r -> M = r + 8*(lane>>4), N = lane&15
#pragma unroll
    for (int r = 0; r < 8; ++r) {
      G[(rt * 16 + hl * 8 + r) * GSTR + (ct0    ) * 16 + m] = acc0[r];
      G[(rt * 16 + hl * 8 + r) * GSTR + (ct0 + 1) * 16 + m] = acc1[r];
    }
    __syncthreads();

    // ---- running top-11 on dist^2 = ||xi||^2 + ||xj||^2 - 2 g ----
    const float* sqj = sqjA + cur * CB;
#pragma unroll
    for (int c16 = 0; c16 < 16; ++c16) {
      int cc = selCol + c16 * 4;
      float d2 = sqi + sqj[cc] - 2.0f * G[selRow * GSTR + cc];
      insert11(best, d2);
    }
    wait_async();                       // prefetched tile has landed in LDS
    __syncthreads();
    cur ^= 1;
  }

  // ---- merge the 4 partial top-11 lists per row ----
  float* mrg = (float*)Bh;              // reuse LDS: 64*4*11*4 = 11264 B
#pragma unroll
  for (int q = 0; q < TOPN; ++q)
    mrg[(selRow * 4 + selCol) * TOPN + q] = best[q];
  __syncthreads();

  if (tid < RB) {
    float fin[TOPN];
#pragma unroll
    for (int q = 0; q < TOPN; ++q) fin[q] = 3.0e38f;
    for (int t2 = 0; t2 < 4; ++t2)
      for (int q = 0; q < TOPN; ++q)
        insert11(fin, mrg[(tid * 4 + t2) * TOPN + q]);
    float ssum = 0.0f;
#pragma unroll
    for (int q = 0; q < TOPN; ++q) ssum += sqrtf(fmaxf(fin[q], 1e-12f));
    ssum -= sqrtf(fmaxf(fin[0], 1e-12f));    // drop self (smallest)
    sumk[rowBase + tid] = ssum;
  }
}

// -------------------- mean of sum_k --------------------------------------
__global__ __launch_bounds__(256) void mean_kernel(const float* __restrict__ sumk,
                                                   float* __restrict__ meanp)
{
  __shared__ float red[256];
  const int t = threadIdx.x;
  float a = 0.0f;
  for (int i = t; i < N_PTS; i += 256) a += sumk[i];
  red[t] = a;
  __syncthreads();
  for (int off = 128; off > 0; off >>= 1) {
    if (t < off) red[t] += red[t + off];
    __syncthreads();
  }
  if (t == 0) meanp[0] = red[0] / (float)N_PTS;
}

// -------------------- elementwise output ----------------------------------
__global__ __launch_bounds__(256) void finalize_kernel(const float* __restrict__ sumk,
                                                       const float* __restrict__ meanp,
                                                       float* __restrict__ out)
{
  const int i = blockIdx.x * 256 + threadIdx.x;
  if (i < N_PTS) {
    const float m  = meanp[0];
    const float sk = sumk[i];
    const float ratio = (sk * sk) / (m * m);
    out[i] = 0.001f / (ratio + 0.001f);
  }
}

extern "C" void kernel_launch(void* const* d_in, const int* in_sizes, int n_in,
                              void* d_out, int out_size, void* d_ws, size_t ws_size,
                              hipStream_t stream) {
  const float* s  = (const float*)d_in[0];
  const float* W1 = (const float*)d_in[1];
  const float* b1 = (const float*)d_in[2];
  const float* W2 = (const float*)d_in[3];
  const float* b2 = (const float*)d_in[4];
  float* out = (float*)d_out;
  (void)in_sizes; (void)n_in; (void)out_size; (void)ws_size;

  char* w = (char*)d_ws;
  _Float16* xh   = (_Float16*)(w);                  // 2 MB   [8192*128]
  _Float16* xl   = (_Float16*)(w + 2097152);        // 2 MB
  _Float16* sh   = (_Float16*)(w + 4194304);        // 8 MB   [8192*512]
  _Float16* sl   = (_Float16*)(w + 12582912);       // 8 MB
  _Float16* w1h  = (_Float16*)(w + 20971520);       // 128 KB [128*512]
  _Float16* w1l  = (_Float16*)(w + 21102592);       // 128 KB
  _Float16* w2h  = (_Float16*)(w + 21233664);       // 32 KB  [128*128]
  _Float16* w2l  = (_Float16*)(w + 21266432);       // 32 KB
  float*    sq   = (float*)   (w + 21299200);       // 32 KB
  float*    sumk = (float*)   (w + 21331968);       // 32 KB
  float*    mnp  = (float*)   (w + 21364736);       // 4 B

  split_kernel<<<(N_PTS * SDIM) / 256, 256, 0, stream>>>(s, sh, sl, N_PTS * SDIM);
  split_kernel<<<(HID * SDIM) / 256, 256, 0, stream>>>(W1, w1h, w1l, HID * SDIM);
  split_kernel<<<(DEMB * HID) / 256, 256, 0, stream>>>(W2, w2h, w2l, DEMB * HID);
  mlp_kernel<<<N_PTS / 64, 256, 0, stream>>>(sh, sl, w1h, w1l, w2h, w2l,
                                             b1, b2, xh, xl, sq);
  knn_kernel<<<N_PTS / RB, 256, 0, stream>>>(xh, xl, sq, sumk);
  mean_kernel<<<1, 256, 0, stream>>>(sumk, mnp);
  finalize_kernel<<<N_PTS / 256, 256, 0, stream>>>(sumk, mnp, out);
}
